// AugmentedNeuralODE_26268019982978
// MI455X (gfx1250) — matrix operand order repaired and checked by reading, not compiled, not run on hardware
//
#include <hip/hip_runtime.h>
#include <hip/hip_bf16.h>

// CDNA5 (gfx1250) wave32 WMMA implementation of an augmented Neural-ODE RK4
// integrator. One block of 8 wave32 per 16-sample tile; weights pre-packed to
// f16 WMMA B-fragment layout; activations staged in LDS in A-fragment layout;
// RK4 state and accumulators live in f32 registers per wave. Branchless
// exp2/rcp tanh keeps EXEC all-ones and co-executes with XDL WMMAs.
// NOTE: keep a single rotating A-fragment per GEMM — wider manual double
// buffering overflows the register budget and spills the weight cache.

typedef __attribute__((ext_vector_type(16))) _Float16 v16h;
typedef __attribute__((ext_vector_type(8)))  float    v8f;

#define NB    2048
#define NT    256
#define DD    128     // augmented state dim (125 data + 3 aug)
#define DDATA 125
#define HH    512

#define W1P (DD*HH)   // 65536  packed f16 elements
#define W2P (HH*HH)   // 262144
#define W3P (HH*DD)   // 65536

__device__ __forceinline__ v8f wmma_f16(v16h a, v16h b, v8f c) {
  // D = A(16x32 f16) * B(32x16 f16) + C(16x16 f32)
  return __builtin_amdgcn_wmma_f32_16x16x32_f16(
      /*neg_a=*/false, a, /*neg_b=*/false, b,
      /*c_mod=*/(short)0, c, /*reuse_a=*/false, /*reuse_b=*/false);
}

// Branchless tanh: 1 - 2/(exp2(2*log2e*x)+1). Saturates to +/-1 at +/-inf,
// no EXEC-modifying slow path, 3 VALU + v_exp_f32 + v_rcp_f32.
__device__ __forceinline__ float fast_tanh(float x) {
  float e = __builtin_amdgcn_exp2f(x * 2.885390081777927f);  // exp(2x)
  return 1.0f - 2.0f * __builtin_amdgcn_rcpf(e + 1.0f);
}

// ---------------------------------------------------------------------------
// Pack row-major f32 weights into f16 WMMA B-fragment blocks.
// Block (nt,kt) covers K=[kt*32,kt*32+32), N=[nt*16,nt*16+16); block index =
// nt*nblkK + kt; within a block lane L holds 16 f16 at byte offset L*32 with
// element e -> K = kt*32 + (e/8)*16 + (L/16)*8 + (e%8), N = nt*16 + (L%16).
// ---------------------------------------------------------------------------
__global__ void pack_weights_kernel(const float* __restrict__ W1,
                                    const float* __restrict__ W2,
                                    const float* __restrict__ W3,
                                    _Float16* __restrict__ ws) {
  int idx = blockIdx.x * blockDim.x + threadIdx.x;
  const int total = W1P + W2P + W3P;
  if (idx >= total) return;
  const float* W; int Ktot, Ntot, local;
  if (idx < W1P)            { W = W1; Ktot = DD; Ntot = HH; local = idx; }
  else if (idx < W1P + W2P) { W = W2; Ktot = HH; Ntot = HH; local = idx - W1P; }
  else                      { W = W3; Ktot = HH; Ntot = DD; local = idx - W1P - W2P; }
  int nblkK  = Ktot >> 5;
  int blk    = local >> 9;      // 512 f16 per block
  int within = local & 511;
  int lane   = within >> 4;
  int e      = within & 15;
  int nt = blk / nblkK;
  int kt = blk % nblkK;
  int K = kt * 32 + ((e >> 3) << 4) + (lane >> 4) * 8 + (e & 7);
  int N = nt * 16 + (lane & 15);
  ws[idx] = (_Float16)W[(size_t)K * Ntot + N];
}

// ---------------------------------------------------------------------------
// Main persistent integrator kernel. grid = 128 blocks (16 samples each),
// block = 256 threads = 8 wave32.
// ---------------------------------------------------------------------------
__global__ __launch_bounds__(256)
void ode_rk4_kernel(const float* __restrict__ x_ivps,   // [B,1,125]
                    const float* __restrict__ t_seg,    // [B,T]
                    const int*   __restrict__ seg_lens, // [B]
                    const float* __restrict__ b1,       // [512]
                    const float* __restrict__ b2,       // [512]
                    const float* __restrict__ b3,       // [128]
                    const _Float16* __restrict__ wp,    // packed weights
                    float* __restrict__ out)            // [B,T,1,128]
{
  // Activations in WMMA A-fragment layout: per K-block of 32, lane L holds
  // 16 f16: M = L%16, K = kt*32 + (e/8)*16 + (L/16)*8 + (e%8).
  __shared__ __align__(128) _Float16 sZin[4  * 512];  //  4 KB: z input, K=128
  __shared__ __align__(128) _Float16 sHA [16 * 512];  // 16 KB: h1, K=512
  __shared__ __align__(128) _Float16 sHB [16 * 512];  // 16 KB: h2, K=512
  __shared__ float sDt[16];

  const _Float16* W1p = wp;
  const _Float16* W2p = wp + W1P;
  const _Float16* W3p = wp + W1P + W2P;

  const int tid = threadIdx.x;
  const int w   = tid >> 5;     // wave id 0..7
  const int L   = tid & 31;     // lane id
  const int hi  = L >> 4;       // C/D row group: rows M = r + hi*8
  const int col = L & 15;       // C/D column within tile
  const int b0  = blockIdx.x * 16;

  // This wave's z-feature column (GEMM3 output / state slice) for this lane.
  const int kkD = w * 16 + col;
  const float b3v = b3[kkD];

  // Precomputed A-layout scatter base for writing column kk of an activation:
  // addr(r) = base + r*16, where
  // base = (kk>>5)*512 + (((kk&31)>>3)&1)*256 + hi*128 + ((((kk&31)>>4)<<3)|(kk&7))
  int offD;
  {
    int r32 = kkD & 31;
    offD = (kkD >> 5) * 512 + (((r32 >> 3) & 1) << 8) + hi * 128 +
           (((r32 >> 4) << 3) | (r32 & 7));
  }
  // Per-tile hidden-layer scatter bases + biases (n constant per thread/tile).
  int   offH[4];
  float bias1[4], bias2[4];
#pragma unroll
  for (int j = 0; j < 4; ++j) {
    int n   = (w * 4 + j) * 16 + col;
    int r32 = n & 31;
    offH[j] = (n >> 5) * 512 + (((r32 >> 3) & 1) << 8) + hi * 128 +
              (((r32 >> 4) << 3) | (r32 & 7));
    bias1[j] = b1[n];
    bias2[j] = b2[n];
  }

  float zb[8];    // z_base[M][kkD], M = r + hi*8
  float accv[8];  // RK4 accumulator k1+2k2+2k3+k4

  // ---- init: z0 = [x_ivps | zeros], emit sol[:,0], seed zin fragments ----
#pragma unroll
  for (int r = 0; r < 8; ++r) {
    int m = r + hi * 8;
    long b = b0 + m;
    float z = (kkD < DDATA) ? x_ivps[b * DDATA + kkD] : 0.0f;
    zb[r] = z;
    out[(b * (long)NT) * DD + kkD] = z;
    sZin[offD + r * 16] = (_Float16)z;
  }
  __syncthreads();

  for (int t = 1; t < NT; ++t) {
    // jagged per-sample dt for this step (0 beyond segment -> identity step)
    if (tid < 16) {
      int b = b0 + tid;
      float d = 0.0f;
      if (t < seg_lens[b]) d = t_seg[b * NT + t] - t_seg[b * NT + t - 1];
      sDt[tid] = d;
    }

    for (int ev = 0; ev < 4; ++ev) {
      __syncthreads();  // zin (and sDt) ready

      // ---------- GEMM1: tanh(zin[16x128] @ W1[128x512] + b1) -> sHA ------
      {
        v8f acc[4] = {};
        v16h a = *(const v16h*)(&sZin[L * 16]);
        for (int kt = 0; kt < 4; ++kt) {
          v16h an = *(const v16h*)(&sZin[((kt + 1) & 3) * 512 + L * 16]);
#pragma unroll
          for (int j = 0; j < 4; ++j) {
            v16h bf = *(const v16h*)(W1p + (size_t)((w * 4 + j) * 4 + kt) * 512 + L * 16);
            acc[j] = wmma_f16(a, bf, acc[j]);
          }
          a = an;
        }
#pragma unroll
        for (int j = 0; j < 4; ++j) {
#pragma unroll
          for (int r = 0; r < 8; ++r) {
            float hv = fast_tanh(acc[j][r] + bias1[j]);
            sHA[offH[j] + r * 16] = (_Float16)hv;
          }
        }
      }
      __syncthreads();

      // ---------- GEMM2: tanh(h1[16x512] @ W2[512x512] + b2) -> sHB -------
      {
        v8f acc[4] = {};
        v16h a = *(const v16h*)(&sHA[L * 16]);
        for (int kt = 0; kt < 16; ++kt) {
          v16h an = *(const v16h*)(&sHA[((kt + 1) & 15) * 512 + L * 16]);
#pragma unroll
          for (int j = 0; j < 4; ++j) {
            v16h bf = *(const v16h*)(W2p + (size_t)((w * 4 + j) * 16 + kt) * 512 + L * 16);
            acc[j] = wmma_f16(a, bf, acc[j]);
          }
          a = an;
        }
#pragma unroll
        for (int j = 0; j < 4; ++j) {
#pragma unroll
          for (int r = 0; r < 8; ++r) {
            float hv = fast_tanh(acc[j][r] + bias2[j]);
            sHB[offH[j] + r * 16] = (_Float16)hv;
          }
        }
      }
      __syncthreads();

      // ---------- GEMM3: k = h2[16x512] @ W3[512x128] + b3; RK4 combine ---
      {
        v8f acc = {};
        v16h a = *(const v16h*)(&sHB[L * 16]);
        for (int kt = 0; kt < 16; ++kt) {
          v16h an = *(const v16h*)(&sHB[((kt + 1) & 15) * 512 + L * 16]);
          v16h bf = *(const v16h*)(W3p + (size_t)(w * 16 + kt) * 512 + L * 16);
          acc = wmma_f16(a, bf, acc);
          a = an;
        }
#pragma unroll
        for (int r = 0; r < 8; ++r) {
          float kval = acc[r] + b3v;
          float dtm  = sDt[r + hi * 8];
          float zi;
          if (ev == 0) {
            accv[r] = kval;               zi = zb[r] + 0.5f * dtm * kval;
          } else if (ev == 1) {
            accv[r] += 2.0f * kval;       zi = zb[r] + 0.5f * dtm * kval;
          } else if (ev == 2) {
            accv[r] += 2.0f * kval;       zi = zb[r] + dtm * kval;
          } else {
            accv[r] += kval;
            float zn = zb[r] + (dtm * (1.0f / 6.0f)) * accv[r];
            zb[r] = zn;
            zi = zn;  // becomes next step's k1 input
            long b = b0 + r + hi * 8;
            out[(b * (long)NT + t) * DD + kkD] = zn;
          }
          sZin[offD + r * 16] = (_Float16)zi;
        }
      }
      // next ev's leading __syncthreads() protects sZin readers
    }
    __syncthreads();  // protect sDt before next step's rewrite
  }
}

extern "C" void kernel_launch(void* const* d_in, const int* in_sizes, int n_in,
                              void* d_out, int out_size, void* d_ws, size_t ws_size,
                              hipStream_t stream) {
  const float* x_ivps   = (const float*)d_in[0];
  const float* t_seg    = (const float*)d_in[1];
  const int*   seg_lens = (const int*)  d_in[2];
  const float* W1 = (const float*)d_in[3];
  const float* b1 = (const float*)d_in[4];
  const float* W2 = (const float*)d_in[5];
  const float* b2 = (const float*)d_in[6];
  const float* W3 = (const float*)d_in[7];
  const float* b3 = (const float*)d_in[8];
  _Float16* wp = (_Float16*)d_ws;   // 768 KB of packed f16 weights

  const int total = W1P + W2P + W3P;
  pack_weights_kernel<<<(total + 255) / 256, 256, 0, stream>>>(W1, W2, W3, wp);
  ode_rk4_kernel<<<NB / 16, 256, 0, stream>>>(x_ivps, t_seg, seg_lens,
                                              b1, b2, b3, wp, (float*)d_out);
}
